// CausalSelfAttention_85959475462660
// MI455X (gfx1250) — matrix operand order repaired and checked by reading, not compiled, not run on hardware
//
#include <hip/hip_runtime.h>
#include <hip/hip_bf16.h>

// ---------------------------------------------------------------------------
// CDNA5 (gfx1250) causal self-attention: all-WMMA bf16 path, f32 accumulate,
// TDM (tensor_load_to_lds) double-buffered GEMM staging, async global->LDS
// copies in the flash-attention inner loop.
// B=4, T=2048, D=1024, H=16, DH=64.
// ---------------------------------------------------------------------------

typedef __attribute__((ext_vector_type(16))) __bf16 v16bf;
typedef __attribute__((ext_vector_type(8)))  float  v8f;
typedef __attribute__((ext_vector_type(4)))  unsigned int u4;
typedef __attribute__((ext_vector_type(4)))  unsigned int u32x4;
typedef __attribute__((ext_vector_type(8)))  int i32x8;
typedef __attribute__((ext_vector_type(4)))  int i32x4;

#define ATT_H 16

#if defined(__has_builtin)
#  if __has_builtin(__builtin_amdgcn_tensor_load_to_lds)
#    define HAVE_TDM 1
#  endif
#endif
#ifndef HAVE_TDM
#  define HAVE_TDM 0
#endif

// LDS byte address: ISA maps LDS-aperture generic addresses via addr[31:0].
__device__ __forceinline__ unsigned lds_off(const void* p) {
  return (unsigned)(unsigned long long)p;
}

// Async global->LDS copies (VGLOBAL async encoding; INST_OFFSET is added to
// both the LDS and global address, so one address pair covers 32 B).
__device__ __forceinline__ void async_cp_b128(unsigned lds,
                                              unsigned long long g) {
  asm volatile("global_load_async_to_lds_b128 %0, %1, off" ::"v"(lds), "v"(g)
               : "memory");
}
__device__ __forceinline__ void async_cp_b128x2(unsigned lds,
                                                unsigned long long g) {
  asm volatile(
      "global_load_async_to_lds_b128 %0, %1, off\n\t"
      "global_load_async_to_lds_b128 %0, %1, off offset:16" ::"v"(lds), "v"(g)
      : "memory");
}
__device__ __forceinline__ void wait_async0() {
  asm volatile("s_wait_asynccnt 0" ::: "memory");
}

// s_wait_tensorcnt takes an immediate -> template parameter.
template <int N>
__device__ __forceinline__ void wait_tensor() {
#if defined(__has_builtin)
#  if __has_builtin(__builtin_amdgcn_s_wait_tensorcnt)
  __builtin_amdgcn_s_wait_tensorcnt(N);
  return;
#  endif
#endif
  if (N == 0)
    asm volatile("s_wait_tensorcnt 0" ::: "memory");
  else
    asm volatile("s_wait_tensorcnt 2" ::: "memory");
}

#if HAVE_TDM
// Issue one 2-D bf16 tile DMA: tile_d1 rows x tile_d0 elems, row pitch
// `rowlen` elems, packed contiguously at `lds`. D# per ISA ch.8.
__device__ __forceinline__ void tdm_issue_2d(const void* g, unsigned lds,
                                             unsigned tile_d0,
                                             unsigned tile_d1,
                                             unsigned rowlen) {
  unsigned long long ga = (unsigned long long)g;
  u32x4 g0;
  g0[0] = 1u;                                   // count=1, user descriptor
  g0[1] = lds;                                  // lds_addr
  g0[2] = (unsigned)ga;                         // global_addr[31:0]
  g0[3] = (unsigned)((ga >> 32) & 0x01FFFFFFu)  // global_addr[56:32]
          | (2u << 30);                         // type=2 ("image")
  i32x8 g1;
  g1[0] = (int)(1u << 16);                      // data_size=1 (2 bytes)
  g1[1] = (int)((rowlen & 0xFFFFu) << 16);      // tensor_dim0[15:0]
  g1[2] = (int)(rowlen >> 16);                  // tensor_dim0[31:16], dim1 lo=0
  g1[3] = (int)(0x10u | (tile_d0 << 16));       // tensor_dim1=1<<20, tile_dim0
  g1[4] = (int)(tile_d1 & 0xFFFFu);             // tile_dim1 (tile_dim2=0)
  g1[5] = (int)rowlen;                          // tensor_dim0_stride[31:0]
  g1[6] = 0;                                    // stride hi / dim1_stride lo
  g1[7] = 0;
  i32x4 z4 = {0, 0, 0, 0};
#if defined(__clang_major__) && __clang_major__ >= 23
  i32x8 z8 = {0, 0, 0, 0, 0, 0, 0, 0};
  __builtin_amdgcn_tensor_load_to_lds(g0, g1, z4, z4, z8, 0);
#else
  __builtin_amdgcn_tensor_load_to_lds(g0, g1, z4, z4, 0);
#endif
}
#endif

// ---------------------------------------------------------------------------
// Elementwise fp32 -> bf16
// ---------------------------------------------------------------------------
__global__ void k_f32_to_bf16(const float* __restrict__ in,
                              __bf16* __restrict__ out, int n) {
  int i = blockIdx.x * blockDim.x + threadIdx.x;
  if (i < n) out[i] = (__bf16)in[i];
}

// in: K x N row-major (fp32) ; out: N x K row-major (bf16)  == B^T
__global__ void k_transpose_to_bf16(const float* __restrict__ in,
                                    __bf16* __restrict__ out, int K, int N) {
  int i = blockIdx.x * blockDim.x + threadIdx.x;
  if (i < K * N) {
    int n = i / K;
    int k = i - n * K;
    out[(size_t)n * K + k] = (__bf16)in[(size_t)k * N + n];
  }
}

// ---------------------------------------------------------------------------
// Tiled GEMM: C(MxN) = A(MxK) * B(KxN), with B given transposed (BT: N x K).
// Block tile 128x64, 8 waves (256 thr), each wave does 16 rows x 64 cols.
// K stepped by 32; double-buffered LDS fed by the Tensor Data Mover.
// ---------------------------------------------------------------------------
template <typename CT>
__global__ __launch_bounds__(256) void gemm_bf16_wmma(
    const __bf16* __restrict__ A, const __bf16* __restrict__ BT,
    CT* __restrict__ C, int M, int N, int K) {
  __shared__ __bf16 Alds[2][128 * 32];
  __shared__ __bf16 Blds[2][64 * 32];

  const int tid = threadIdx.x;
  const int lane = tid & 31;
  const int w = tid >> 5;       // wave 0..7
  const int lg = lane >> 4;     // lane group (0/1)
  const int ln = lane & 15;
  const int mBase = blockIdx.y * 128;
  const int nBase = blockIdx.x * 64;

  v8f acc0 = {}, acc1 = {}, acc2 = {}, acc3 = {};

  union frag { v16bf v; u4 u[2]; };

  auto compute = [&](const __bf16* Ab, const __bf16* Bb) {
    // A fragment (16x32 bf16): lane holds row (ln), kbase = lg*8 and +16.
    frag af;
    const __bf16* ap = Ab + (w * 16 + ln) * 32 + lg * 8;
    af.u[0] = *(const u4*)(ap);
    af.u[1] = *(const u4*)(ap + 16);
    // Four B fragments (32x16 each): lane holds column, kbase = lg*8 and +16.
    const __bf16* bp = Bb + ln * 32 + lg * 8;
#define GEMM_TILE(i, accv)                                                   \
  {                                                                          \
    frag bf;                                                                 \
    const __bf16* p = bp + (i) * 16 * 32;                                    \
    bf.u[0] = *(const u4*)(p);                                               \
    bf.u[1] = *(const u4*)(p + 16);                                          \
    accv = __builtin_amdgcn_wmma_f32_16x16x32_bf16(false, af.v, false, bf.v, \
                                                   (short)0, accv, false,    \
                                                   false);                   \
  }
    GEMM_TILE(0, acc0)
    GEMM_TILE(1, acc1)
    GEMM_TILE(2, acc2)
    GEMM_TILE(3, acc3)
#undef GEMM_TILE
  };

  const int nk = K / 32;

#if HAVE_TDM
  const unsigned la[2] = {lds_off(&Alds[0][0]), lds_off(&Alds[1][0])};
  const unsigned lb[2] = {lds_off(&Blds[0][0]), lds_off(&Blds[1][0])};
  auto issue = [&](int i, int buf) {
    tdm_issue_2d(A + (size_t)mBase * K + i * 32, la[buf], 32u, 128u,
                 (unsigned)K);
    tdm_issue_2d(BT + (size_t)nBase * K + i * 32, lb[buf], 32u, 64u,
                 (unsigned)K);
  };
  if (w == 0) issue(0, 0);
  for (int i = 0; i < nk; ++i) {
    const int cur = i & 1;
    if (w == 0) {
      if (i + 1 < nk) {
        issue(i + 1, cur ^ 1);
        wait_tensor<2>();   // tile i's two DMAs complete (in-order)
      } else {
        wait_tensor<0>();
      }
    }
    __syncthreads();
    compute(Alds[cur], Blds[cur]);
    __syncthreads();
  }
#else
  // Fallback: async global->LDS copies, single buffer.
  for (int i = 0; i < nk; ++i) {
    const int k0 = i * 32;
    __syncthreads();
    {
      int r = tid >> 1;
      int c = (tid & 1) * 16;
      async_cp_b128x2(lds_off(&Alds[0][r * 32 + c]),
                      (unsigned long long)(A + (size_t)(mBase + r) * K + k0 +
                                           c));
      int n = tid >> 2;
      int c2 = (tid & 3) * 8;
      async_cp_b128(lds_off(&Blds[0][n * 32 + c2]),
                    (unsigned long long)(BT + (size_t)(nBase + n) * K + k0 +
                                         c2));
    }
    wait_async0();
    __syncthreads();
    compute(Alds[0], Blds[0]);
  }
#endif

  // C layout: m = vgpr + 8*lanegroup, n = lane&15 (+16 per tile).
  const int m0 = mBase + w * 16 + lg * 8;
  const int nc = nBase + ln;
#pragma unroll
  for (int v = 0; v < 8; ++v) {
    CT* cp = C + (size_t)(m0 + v) * N + nc;
    cp[0]  = (CT)acc0[v];
    cp[16] = (CT)acc1[v];
    cp[32] = (CT)acc2[v];
    cp[48] = (CT)acc3[v];
  }
}

// ---------------------------------------------------------------------------
// Flash attention over one (b,h) and a 64-row Q tile. 4 waves / 128 threads.
// qkv: (B*T) x 3D bf16 rows; out: (B*T) x D bf16.
// ---------------------------------------------------------------------------
__global__ __launch_bounds__(128) void attn_fwd(
    const __bf16* __restrict__ qkv, __bf16* __restrict__ out, int T, int D3,
    int D) {
  __shared__ __bf16 Klds[32 * 64];      // K tile, row-major (key x dh)
  __shared__ __bf16 Vt[64 * 32];        // V tile, transposed (dh x key)
  __shared__ __bf16 Pl[4 * 16 * 32];    // per-wave P round-trip

  const int tid = threadIdx.x;
  const int lane = tid & 31;
  const int w = tid >> 5;
  const int lg = lane >> 4;
  const int ln = lane & 15;

  const int nQT = T / 64;
  const int qt = blockIdx.x % nQT;
  const int bh = blockIdx.x / nQT;
  const int h = bh % ATT_H;
  const int b = bh / ATT_H;
  const int qBase = qt * 64;
  const int hcol = h * 64;
  const __bf16* base = qkv + (size_t)b * T * D3;

  union frag { v16bf v; u4 u[2]; };

  // Preload Q A-fragments for both K-steps (d 0..31, 32..63).
  frag qa0, qa1;
  {
    const __bf16* qp =
        base + (size_t)(qBase + w * 16 + ln) * D3 + hcol + lg * 8;
    qa0.u[0] = *(const u4*)(qp);
    qa0.u[1] = *(const u4*)(qp + 16);
    qa1.u[0] = *(const u4*)(qp + 32);
    qa1.u[1] = *(const u4*)(qp + 48);
  }

  v8f oacc[4] = {{}, {}, {}, {}};
  float mrun[8], lrun[8];
#pragma unroll
  for (int v = 0; v < 8; ++v) {
    mrun[v] = -3.0e38f;
    lrun[v] = 0.0f;
  }

  const int nKV = qBase / 32 + 2;  // causal bound: keys <= qBase+63
  for (int j = 0; j < nKV; ++j) {
    const int kvBase = j * 32;
    __syncthreads();
    // Stage K (32x64) via async global->LDS; V transposed via VGPR path.
    {
      int r = tid >> 2;
      int c = (tid & 3) * 16;
      const __bf16* kp = base + (size_t)(kvBase + r) * D3 + D + hcol + c;
      async_cp_b128x2(lds_off(&Klds[r * 64 + c]), (unsigned long long)kp);
      if (j + 1 < nKV)  // global_prefetch for next KV tile
        __builtin_prefetch(kp + (size_t)32 * D3, 0, 0);
      const __bf16* vp = base + (size_t)(kvBase + r) * D3 + 2 * D + hcol + c;
#pragma unroll
      for (int i = 0; i < 16; ++i) Vt[(c + i) * 32 + r] = vp[i];
    }
    wait_async0();
    __syncthreads();

    // S = Q * K^T : two 16x16 S-tiles, each over 2 K-steps of 32.
    v8f sacc[2] = {{}, {}};
#pragma unroll
    for (int t = 0; t < 2; ++t) {
      const __bf16* kb = Klds + (t * 16 + ln) * 64 + lg * 8;
      frag bf0, bf1;
      bf0.u[0] = *(const u4*)(kb);
      bf0.u[1] = *(const u4*)(kb + 16);
      bf1.u[0] = *(const u4*)(kb + 32);
      bf1.u[1] = *(const u4*)(kb + 48);
      sacc[t] = __builtin_amdgcn_wmma_f32_16x16x32_bf16(
          false, qa0.v, false, bf0.v, (short)0, sacc[t], false, false);
      sacc[t] = __builtin_amdgcn_wmma_f32_16x16x32_bf16(
          false, qa1.v, false, bf1.v, (short)0, sacc[t], false, false);
    }

    // Scale, causal mask, online softmax (row = vgpr + 8*lanegroup).
    float pv0[8], pv1[8];
#pragma unroll
    for (int v = 0; v < 8; ++v) {
      const int qi = qBase + w * 16 + lg * 8 + v;
      float s0 = sacc[0][v] * 0.125f;
      float s1 = sacc[1][v] * 0.125f;
      if (kvBase + ln > qi) s0 = -3.0e38f;
      if (kvBase + 16 + ln > qi) s1 = -3.0e38f;
      float tm = fmaxf(s0, s1);
      for (int off = 1; off < 16; off <<= 1)
        tm = fmaxf(tm, __shfl_xor(tm, off, 32));
      float mnew = fmaxf(mrun[v], tm);
      float alpha = __expf(mrun[v] - mnew);
      float p0 = __expf(s0 - mnew);
      float p1 = __expf(s1 - mnew);
      float ps = p0 + p1;
      for (int off = 1; off < 16; off <<= 1) ps += __shfl_xor(ps, off, 32);
      lrun[v] = lrun[v] * alpha + ps;
      mrun[v] = mnew;
#pragma unroll
      for (int t = 0; t < 4; ++t) oacc[t][v] *= alpha;
      pv0[v] = p0;
      pv1[v] = p1;
    }

    // Re-layout P (accum layout -> A-fragment layout) via per-wave LDS.
    {
      __bf16* pw = Pl + w * 16 * 32;
#pragma unroll
      for (int v = 0; v < 8; ++v) {
        int m = lg * 8 + v;
        pw[m * 32 + ln] = (__bf16)pv0[v];
        pw[m * 32 + 16 + ln] = (__bf16)pv1[v];
      }
    }
    __syncthreads();

    // O += P * V  (K = 32 keys, four 16-wide dh tiles).
    frag pa;
    {
      const __bf16* pp = Pl + w * 16 * 32 + ln * 32 + lg * 8;
      pa.u[0] = *(const u4*)(pp);
      pa.u[1] = *(const u4*)(pp + 16);
    }
#pragma unroll
    for (int t = 0; t < 4; ++t) {
      frag bf;
      const __bf16* vbp = Vt + (t * 16 + ln) * 32 + lg * 8;
      bf.u[0] = *(const u4*)(vbp);
      bf.u[1] = *(const u4*)(vbp + 16);
      oacc[t] = __builtin_amdgcn_wmma_f32_16x16x32_bf16(
          false, pa.v, false, bf.v, (short)0, oacc[t], false, false);
    }
  }

  // Epilogue: O / l -> bf16 out at (b, t, h*64 + dh).
  __bf16* op = out + (size_t)b * T * D;
#pragma unroll
  for (int v = 0; v < 8; ++v) {
    float inv = 1.0f / lrun[v];
    int row = qBase + w * 16 + lg * 8 + v;
    __bf16* orow = op + (size_t)row * D + hcol + ln;
    orow[0]  = (__bf16)(oacc[0][v] * inv);
    orow[16] = (__bf16)(oacc[1][v] * inv);
    orow[32] = (__bf16)(oacc[2][v] * inv);
    orow[48] = (__bf16)(oacc[3][v] * inv);
  }
}

// ---------------------------------------------------------------------------
// Launcher
// ---------------------------------------------------------------------------
extern "C" void kernel_launch(void* const* d_in, const int* in_sizes, int n_in,
                              void* d_out, int out_size, void* d_ws,
                              size_t ws_size, hipStream_t stream) {
  (void)in_sizes; (void)n_in; (void)out_size; (void)ws_size;
  const float* x    = (const float*)d_in[0];
  // d_in[1] = causal mask (analytic, unused)
  const float* Wqkv = (const float*)d_in[2];
  const float* Wout = (const float*)d_in[3];
  float* out = (float*)d_out;

  const int B = 4, T = 2048, D = 1024;
  const int M = B * T;       // 8192
  const int D3 = 3 * D;      // 3072

  char* ws = (char*)d_ws;
  __bf16* xb    = (__bf16*)ws;                          // M*D   (also attn out)
  __bf16* wqkvT = (__bf16*)(ws + (size_t)M * D * 2);    // D3*D
  __bf16* woutT = wqkvT + (size_t)D3 * D;               // D*D
  __bf16* qkvb  = woutT + (size_t)D * D;                // M*D3

  // Convert inputs to bf16 (weights transposed for B-fragment-friendly LDS).
  k_f32_to_bf16<<<(M * D + 255) / 256, 256, 0, stream>>>(x, xb, M * D);
  k_transpose_to_bf16<<<(D * D3 + 255) / 256, 256, 0, stream>>>(Wqkv, wqkvT,
                                                                D, D3);
  k_transpose_to_bf16<<<(D * D + 255) / 256, 256, 0, stream>>>(Wout, woutT,
                                                               D, D);

  // QKV projection: (M x D) * (D x 3D) -> bf16 qkv.
  dim3 g1(D3 / 64, M / 128);
  gemm_bf16_wmma<__bf16><<<g1, 256, 0, stream>>>(xb, wqkvT, qkvb, M, D3, D);

  // Flash attention; writes bf16 (B,T,D) into xb (x no longer needed).
  attn_fwd<<<(B * ATT_H) * (T / 64), 128, 0, stream>>>(qkvb, xb, T, D3, D);

  // Output projection: (M x D) * (D x D) -> fp32 d_out.
  dim3 g2(D / 64, M / 128);
  gemm_bf16_wmma<float><<<g2, 256, 0, stream>>>(xb, woutT, out, M, D, D);
}
